// KronLinear_45921790329283
// MI455X (gfx1250) — matrix operand order repaired and checked by reading, not compiled, not run on hardware
//
#include <hip/hip_runtime.h>

// ---------------------------------------------------------------------------
// KronLinear on MI455X (gfx1250, wave32):
//   Phase 1: build w^T = (sum_r kron(a_r, b_r))^T as bf16 hi/lo split (in d_ws)
//   Phase 2: 8192x4096x4096 GEMM, v_wmma_f32_16x16x32_bf16 with 3-term bf16
//            split (xh*wh + xl*wh + xh*wl), fp32 accumulation.
//            w-tiles streamed into LDS by the Tensor Data Mover (TDM),
//            double-buffered, synchronized with s_wait_tensorcnt.
// ---------------------------------------------------------------------------

typedef __attribute__((ext_vector_type(16))) __bf16 v16bf;
typedef __attribute__((ext_vector_type(8)))  float  v8f;
typedef unsigned int u32x4 __attribute__((ext_vector_type(4)));
typedef int          i32x8 __attribute__((ext_vector_type(8)));
typedef int          i32x4 __attribute__((ext_vector_type(4)));

#define TOKENS 8192
#define DIM    4096   // K == N
#define RANKS  64

#define BM 128
#define BN 128
#define BK 32
#define LDT 40        // LDS row stride in bf16 (80B = 64B data + 16B pad)
#define KSTEPS (DIM / BK)

#if __has_builtin(__builtin_amdgcn_tensor_load_to_lds)
#define USE_TDM 1
#else
#define USE_TDM 0
#endif

union Frag {
  v16bf v;
  uint4 q[2];
};

// ---------------------------------------------------------------------------
// Phase 1: wt[n][kk] = sum_r a[r,i,j]*b[r,k,l]  (kk=i*64+k, n=j*64+l), N-major.
// ---------------------------------------------------------------------------
__global__ __launch_bounds__(256) void build_wt(const float* __restrict__ a,
                                                const float* __restrict__ b,
                                                __bf16* __restrict__ wt_hi,
                                                __bf16* __restrict__ wt_lo) {
  unsigned gid = blockIdx.x * 256u + threadIdx.x;
  unsigned kk = gid & (DIM - 1u);
  unsigned n  = gid >> 12;
  unsigned i = kk >> 6, k = kk & 63u;
  unsigned j = n  >> 6, l = n  & 63u;

  const float* ap = a + i * 64u + j;   // stride DIM over r
  const float* bp = b + k * 64u + l;

  float acc = 0.0f;
#pragma unroll 8
  for (int r = 0; r < RANKS; ++r)
    acc = fmaf(ap[(size_t)r * DIM], bp[(size_t)r * DIM], acc);

  __bf16 hi = (__bf16)acc;
  __bf16 lo = (__bf16)(acc - (float)hi);
  wt_hi[(size_t)n * DIM + kk] = hi;
  wt_lo[(size_t)n * DIM + kk] = lo;
}

#if USE_TDM
// TDM: DMA a BN x BK bf16 tile (row stride DIM elements) into LDS with
// 16B-per-row padding so rows land at LDT(=40)-element pitch.
// D# per CDNA5 ISA ch.8: group0 = {count/type, lds_addr, global_addr},
// group1 = {flags, dims, tile dims, strides}; groups 2/3 zero (2D tensor).
__device__ __forceinline__ void tdm_load_wt(const __bf16* gptr, unsigned lds_off) {
  unsigned long long ga = (unsigned long long)(uintptr_t)gptr;
  u32x4 g0;
  g0[0] = 1u;                                     // count=1, user desc
  g0[1] = lds_off;                                // lds_addr (bytes)
  g0[2] = (unsigned)ga;                           // global_addr[31:0]
  g0[3] = ((unsigned)(ga >> 32) & 0x01FFFFFFu)    // global_addr[56:32]
          | (2u << 30);                           // type=2 ("image")
  i32x8 g1;
  g1[0] = (int)((1u << 16)      // data_size = 2 bytes
              | (1u << 20)      // pad_enable
              | (3u << 22)      // pad_interval: 16 DWORDs (64B) of data
              | (3u << 25));    // pad_amount:    4 DWORDs (16B) of pad
  g1[1] = (int)((unsigned)DIM << 16);   // tensor_dim0[15:0] in [31:16]
  g1[2] = (int)((unsigned)DIM << 16);   // tensor_dim1[15:0] in [31:16]
  g1[3] = (int)((unsigned)BK << 16);    // tile_dim0 = 32 (K run)
  g1[4] = (int)BN;                      // tile_dim1 = 128 rows, tile_dim2 = 0
  g1[5] = (int)DIM;                     // tensor_dim0_stride (elements)
  g1[6] = 0;
  g1[7] = 0;
  i32x4 z4 = {0, 0, 0, 0};
  i32x8 z8 = {0, 0, 0, 0, 0, 0, 0, 0};
  __builtin_amdgcn_tensor_load_to_lds(g0, g1, z4, z4, z8, 0);
}
#endif

// ---------------------------------------------------------------------------
// Phase 2 GEMM. 256 threads = 8 wave32s in 2(M) x 4(N); wave owns 64x32.
// ---------------------------------------------------------------------------
__global__ __launch_bounds__(256) void kron_gemm(const float* __restrict__ x,
                                                 const __bf16* __restrict__ wt_hi,
                                                 const __bf16* __restrict__ wt_lo,
                                                 const float* __restrict__ bias,
                                                 float* __restrict__ out) {
  __shared__ __align__(16) __bf16 lds_xh[BM * LDT];
  __shared__ __align__(16) __bf16 lds_xl[BM * LDT];
  __shared__ __align__(16) __bf16 lds_wh[2][BN * LDT];
  __shared__ __align__(16) __bf16 lds_wl[2][BN * LDT];

  const unsigned tid   = threadIdx.x;
  const unsigned lane  = tid & 31u;
  const unsigned wave  = tid >> 5;        // 0..7
  const unsigned wm    = wave >> 2;       // 0..1  -> 64-row slab
  const unsigned wn    = wave & 3u;       // 0..3  -> 32-col slab
  const unsigned m0    = blockIdx.y * BM;
  const unsigned n0    = blockIdx.x * BN;
  const unsigned lhalf = lane & 15u;
  const unsigned kg    = lane >> 4;

  v8f acc[4][2];
#pragma unroll
  for (int ms = 0; ms < 4; ++ms)
#pragma unroll
    for (int ns = 0; ns < 2; ++ns)
#pragma unroll
      for (int v = 0; v < 8; ++v) acc[ms][ns][v] = 0.0f;

#if USE_TDM
  // Prime the pipeline: k-step 0 tiles into buffer 0.
  if (wave == 0)
    tdm_load_wt(wt_hi + (size_t)n0 * DIM, (unsigned)(uintptr_t)&lds_wh[0][0]);
  if (wave == 1)
    tdm_load_wt(wt_lo + (size_t)n0 * DIM, (unsigned)(uintptr_t)&lds_wl[0][0]);
#endif

  for (unsigned kt = 0; kt < KSTEPS; ++kt) {
    const unsigned k0 = kt * BK;
#if USE_TDM
    const unsigned cur = kt & 1u;
#else
    const unsigned cur = 0u;
#endif

    // ---- stage x tile: fp32 -> bf16 hi/lo into LDS (1024 float4 / block) ----
#pragma unroll
    for (int it = 0; it < 4; ++it) {
      unsigned idx = tid + it * 256u;        // 0..1023
      unsigned row = idx >> 3;               // 8 float4 per 32-wide row
      unsigned c4  = idx & 7u;
      const float* gx = x + (size_t)(m0 + row) * DIM + k0 + c4 * 4u;
      float4 vx = *(const float4*)gx;
      if (kt + 1 < KSTEPS)
        __builtin_prefetch(gx + BK, 0, 1);   // global_prefetch_b8 next K-step
      __bf16* ph = &lds_xh[row * LDT + c4 * 4u];
      __bf16* pl = &lds_xl[row * LDT + c4 * 4u];
      float f[4] = {vx.x, vx.y, vx.z, vx.w};
#pragma unroll
      for (int e = 0; e < 4; ++e) {
        __bf16 h = (__bf16)f[e];
        ph[e] = h;
        pl[e] = (__bf16)(f[e] - (float)h);
      }
    }

#if USE_TDM
    // ---- issue next k-step's w tiles into the other buffer, then wait so the
    //      current tiles are complete (next may remain in flight). ----
    if (kt + 1 < KSTEPS) {
      const unsigned nxt = cur ^ 1u;
      if (wave == 0)
        tdm_load_wt(wt_hi + (size_t)n0 * DIM + (k0 + BK),
                    (unsigned)(uintptr_t)&lds_wh[nxt][0]);
      if (wave == 1)
        tdm_load_wt(wt_lo + (size_t)n0 * DIM + (k0 + BK),
                    (unsigned)(uintptr_t)&lds_wl[nxt][0]);
      __builtin_amdgcn_s_wait_tensorcnt(1);
    } else {
      __builtin_amdgcn_s_wait_tensorcnt(0);
    }
#else
    // ---- fallback staging: bf16 uint4 copies through VGPRs ----
#pragma unroll
    for (int it = 0; it < 2; ++it) {
      unsigned idx = tid + it * 256u;        // 0..511
      unsigned row = idx >> 2;
      unsigned c   = idx & 3u;               // 8 bf16 per uint4
      size_t   g   = (size_t)(n0 + row) * DIM + k0 + c * 8u;
      *(uint4*)&lds_wh[0][row * LDT + c * 8u] = *(const uint4*)(wt_hi + g);
      *(uint4*)&lds_wl[0][row * LDT + c * 8u] = *(const uint4*)(wt_lo + g);
    }
#endif
    __syncthreads();

    // ---- B fragments (32x16 bf16): contiguous 32B K-run per lane ----
    Frag bh[2], bl[2];
#pragma unroll
    for (int ns = 0; ns < 2; ++ns) {
      unsigned base = (wn * 32u + ns * 16u + lhalf) * LDT + kg * 16u;
      bh[ns].q[0] = *(uint4*)&lds_wh[cur][base];
      bh[ns].q[1] = *(uint4*)&lds_wh[cur][base + 8u];
      bl[ns].q[0] = *(uint4*)&lds_wl[cur][base];
      bl[ns].q[1] = *(uint4*)&lds_wl[cur][base + 8u];
    }

    // ---- A fragments + 3-term split WMMA ----
#pragma unroll
    for (int ms = 0; ms < 4; ++ms) {
      unsigned abase = (wm * 64u + ms * 16u + lhalf) * LDT + kg * 16u;
      Frag ah, al;
      ah.q[0] = *(uint4*)&lds_xh[abase];
      ah.q[1] = *(uint4*)&lds_xh[abase + 8u];
      al.q[0] = *(uint4*)&lds_xl[abase];
      al.q[1] = *(uint4*)&lds_xl[abase + 8u];
#pragma unroll
      for (int ns = 0; ns < 2; ++ns) {
        acc[ms][ns] = __builtin_amdgcn_wmma_f32_16x16x32_bf16(
            false, ah.v, false, bh[ns].v, (short)0, acc[ms][ns], false, false);
        acc[ms][ns] = __builtin_amdgcn_wmma_f32_16x16x32_bf16(
            false, al.v, false, bh[ns].v, (short)0, acc[ms][ns], false, false);
        acc[ms][ns] = __builtin_amdgcn_wmma_f32_16x16x32_bf16(
            false, ah.v, false, bl[ns].v, (short)0, acc[ms][ns], false, false);
      }
    }
    __syncthreads();
  }

  // ---- epilogue: C layout = lanes 0-15 rows m..m+7, lanes 16-31 rows m+8..m+15
#pragma unroll
  for (int ms = 0; ms < 4; ++ms) {
#pragma unroll
    for (int ns = 0; ns < 2; ++ns) {
      unsigned col = n0 + wn * 32u + ns * 16u + lhalf;
      float bs = bias[col];
      unsigned rowb = m0 + wm * 64u + ms * 16u + kg * 8u;
#pragma unroll
      for (int v = 0; v < 8; ++v)
        out[(size_t)(rowb + v) * DIM + col] = acc[ms][ns][v] + bs;
    }
  }
}

// ---------------------------------------------------------------------------
extern "C" void kernel_launch(void* const* d_in, const int* in_sizes, int n_in,
                              void* d_out, int out_size, void* d_ws, size_t ws_size,
                              hipStream_t stream) {
  const float* x    = (const float*)d_in[0];
  const float* a    = (const float*)d_in[1];
  const float* b    = (const float*)d_in[2];
  const float* bias = (const float*)d_in[3];
  float* out = (float*)d_out;

  // workspace: wt_hi (32MB) + wt_lo (32MB), bf16, 4096x4096 each (N-major)
  __bf16* wt_hi = (__bf16*)d_ws;
  __bf16* wt_lo = wt_hi + (size_t)DIM * DIM;

  build_wt<<<(DIM * DIM) / 256, 256, 0, stream>>>(a, b, wt_hi, wt_lo);

  dim3 grid(DIM / BN, TOKENS / BM);
  kron_gemm<<<grid, dim3(256), 0, stream>>>(x, wt_hi, wt_lo, bias, out);
}
